// FFTAttention_18442589569217
// MI455X (gfx1250) — matrix operand order repaired and checked by reading, not compiled, not run on hardware
//
#include <hip/hip_runtime.h>

typedef __attribute__((ext_vector_type(16))) __bf16 v16bf;
typedef __attribute__((ext_vector_type(8)))  float  v8f;

#define BATCH 4
#define CDIM  192
#define IMGW  256
#define HW    65536           // 256*256
#define NT16  (HW / 16)       // 4096 column tiles of 16
#define KSTEPS (CDIM / 32)    // 6 k-steps of 32

// =====================================================================
// Kernel 0: pack conv1x1 weights (f32 -> bf16) into WMMA A-fragment order.
// out[((ot*6+ks)*32 + lane)*16 + i], rows >= 192 come from w1.
// =====================================================================
__global__ __launch_bounds__(256) void pack_weights_kernel(
    const float* __restrict__ w0,   // rows 0..191   (OC x 192 row-major)
    const float* __restrict__ w1,   // rows 192..575 (for fused qkv)
    int octiles,
    __bf16* __restrict__ out)
{
    int idx = blockIdx.x * 256 + threadIdx.x;
    const int i    = idx & 15;
    const int lane = (idx >> 4) & 31;
    int rest = idx >> 9;
    const int ks = rest % KSTEPS;
    const int ot = rest / KSTEPS;
    if (ot >= octiles) return;
    const int lo = lane & 15, hi = lane >> 4;
    const int ocrow = ot * 16 + lo;
    // A (16x32 bf16) layout: elem i -> K=i, hi lanes +16
    const int ka = ks * 32 + i + (hi << 4);
    float v = (ocrow < CDIM) ? w0[(size_t)ocrow * CDIM + ka]
                             : w1[(size_t)(ocrow - CDIM) * CDIM + ka];
    out[idx] = (__bf16)v;
}

// =====================================================================
// Kernel 1: fused conv1x1 q+kv, OC=576.  One block per 16-column tile:
// X panel (192x16) staged once in LDS as bf16 (transposed: xs[n*192+k]),
// 8 waves sweep all 36 OC tiles -> X read from HBM exactly once.
// =====================================================================
__global__ __launch_bounds__(256) void gemm_qkv_wmma(
    const float* __restrict__ x,      // (4,192,HW)
    const __bf16* __restrict__ wpk,   // packed 36*6*32*16
    float* __restrict__ qpre,         // (4,192,HW)
    float* __restrict__ kvpre)        // (4,384,HW)
{
    __shared__ __attribute__((aligned(32))) __bf16 xs[16 * CDIM];
    const int bid = blockIdx.x;             // 4*4096 blocks
    const int nt = bid & (NT16 - 1);
    const int b  = bid >> 12;
    const int n0 = nt * 16;

    const float* xb = x + (size_t)b * CDIM * HW;
#pragma unroll
    for (int it = 0; it < 12; ++it) {       // 3072 elems / 256 threads
        int linear = it * 256 + threadIdx.x;
        int k = linear >> 4, n = linear & 15;
        xs[n * CDIM + k] = (__bf16)xb[(size_t)k * HW + n0 + n];
    }
    __syncthreads();

    const int lane = threadIdx.x & 31;
    const int wave = threadIdx.x >> 5;
    const int lo = lane & 15, hi = lane >> 4;
    const int n = n0 + lo;
    const v16bf* bsrc = (const v16bf*)(xs + lo * CDIM + hi * 16);  // +ks*32 elems

#pragma unroll
    for (int j = 0; j < 5; ++j) {
        const int ot = wave + 8 * j;        // wave-uniform guard: EXEC stays full
        if (ot < 36) {
            v8f acc = {};
#pragma unroll
            for (int ks = 0; ks < KSTEPS; ++ks) {
                v16bf afrag = *(const v16bf*)(wpk + ((size_t)(ot * KSTEPS + ks) * 32 + lane) * 16);
                v16bf bfrag = bsrc[ks * 2];             // ks*32 bf16 = ks*2 v16bf
                acc = __builtin_amdgcn_wmma_f32_16x16x32_bf16(
                          false, afrag, false, bfrag, (short)0, acc, false, false);
            }
#pragma unroll
            for (int r = 0; r < 8; ++r) {
                int oc = ot * 16 + (hi << 3) + r;
                if (oc < CDIM)
                    qpre [(size_t)(b * CDIM + oc) * HW + n] = acc[r];
                else
                    kvpre[(size_t)(b * 2 * CDIM + (oc - CDIM)) * HW + n] = acc[r];
            }
        }
    }
}

// =====================================================================
// Kernel 2/3: depthwise conv (KS=3 pad 1 for q, KS=7 pad 3 for kv)
// =====================================================================
template <int KS, int PAD, int NCH>
__global__ __launch_bounds__(256) void dwconv_kernel(
    const float* __restrict__ in,   // (4,NCH,256,256)
    const float* __restrict__ wdw,  // (NCH,1,KS,KS)
    const float* __restrict__ bdw,  // (NCH)
    float* __restrict__ out)
{
    int idx = blockIdx.x * 256 + threadIdx.x;
    const int wv = idx & (IMGW - 1);
    int rest = idx >> 8;
    const int h = rest & (IMGW - 1);
    rest >>= 8;
    const int c = rest % NCH;
    const int b = rest / NCH;

    const float* base = in + (size_t)(b * NCH + c) * HW;
    const float* wk   = wdw + (size_t)c * KS * KS;
    float acc = bdw[c];
#pragma unroll
    for (int dy = 0; dy < KS; ++dy) {
        int yy = h + dy - PAD;
        if (yy < 0 || yy >= IMGW) continue;
#pragma unroll
        for (int dx = 0; dx < KS; ++dx) {
            int xx = wv + dx - PAD;
            if (xx < 0 || xx >= IMGW) continue;
            acc += wk[dy * KS + dx] * base[yy * IMGW + xx];
        }
    }
    out[idx] = acc;
}

// =====================================================================
// Kernel 4: per-8x8-patch 2-D circular convolution (== irfft2(rfft2*rfft2))
// out[s] = sum_t q[t] * k[(s-t) mod 8 (2D)], per (b, c, patch).
// =====================================================================
__global__ __launch_bounds__(256) void patch_circconv_kernel(
    const float* __restrict__ qdw,   // (4,192,256,256)
    const float* __restrict__ kvdw,  // (4,384,256,256): k = first 192 ch
    float* __restrict__ outm)        // (4,192,256,256)
{
    __shared__ float sq[4][64];
    __shared__ float sk[4][64];
    const int slot = threadIdx.x >> 6;   // 0..3
    const int t    = threadIdx.x & 63;

    int P = blockIdx.x * 4 + slot;       // 786432 patches total, exact
    const int pw = P & 31;  P >>= 5;
    const int ph = P & 31;  P >>= 5;
    const int c  = P % CDIM;
    const int b  = P / CDIM;

    const int ly = t >> 3, lx = t & 7;
    const size_t pix = (size_t)(ph * 8 + ly) * IMGW + (pw * 8 + lx);
    const size_t qoff = (size_t)(b * CDIM + c) * HW + pix;
    const size_t koff = (size_t)(b * 2 * CDIM + c) * HW + pix;
    sq[slot][t] = qdw[qoff];
    sk[slot][t] = kvdw[koff];
    __syncthreads();

    float acc = 0.f;
#pragma unroll
    for (int ty = 0; ty < 8; ++ty) {
        const int ry = (ly - ty) & 7;
#pragma unroll
        for (int tx = 0; tx < 8; ++tx) {
            acc += sq[slot][ty * 8 + tx] * sk[slot][ry * 8 + ((lx - tx) & 7)];
        }
    }
    outm[qoff] = acc;
}

// =====================================================================
// Kernel 5: channel LayerNorm (mean/var over c=192) + affine + gate by v
// =====================================================================
__global__ __launch_bounds__(256) void ln_vmul_kernel(
    const float* __restrict__ mixin,  // (4,192,256,256)
    const float* __restrict__ kvdw,   // v = channels 192..383
    const float* __restrict__ lnw,
    const float* __restrict__ lnb,
    float* __restrict__ out)          // (4,192,256,256)
{
    const int pix = blockIdx.x * 256 + threadIdx.x;   // 262144 pixels, exact
    const int b  = pix >> 16;
    const int hw = pix & (HW - 1);
    const float* e = mixin + (size_t)(b * CDIM) * HW + hw;

    float s = 0.f, s2 = 0.f;
    for (int c = 0; c < CDIM; ++c) {
        float v = e[(size_t)c * HW];
        s += v; s2 += v * v;
    }
    const float mu  = s * (1.f / CDIM);
    const float var = s2 * (1.f / CDIM) - mu * mu;
    const float inv = rsqrtf(var + 1e-5f);

    for (int c = 0; c < CDIM; ++c) {
        float v = e[(size_t)c * HW];
        float y = (v - mu) * inv * lnw[c] + lnb[c];
        y *= kvdw[(size_t)(b * 2 * CDIM + CDIM + c) * HW + hw];
        out[(size_t)(b * CDIM + c) * HW + hw] = y;
    }
}

// =====================================================================
// Kernel 6: final conv1x1 projection (192x192) + bias, same LDS scheme
// =====================================================================
__global__ __launch_bounds__(256) void gemm_proj_wmma(
    const float* __restrict__ src,    // (4,192,HW)
    const __bf16* __restrict__ wpk,   // packed 12*6*32*16
    const float* __restrict__ bproj,  // (192)
    float* __restrict__ out)          // (4,192,HW)
{
    __shared__ __attribute__((aligned(32))) __bf16 xs[16 * CDIM];
    const int bid = blockIdx.x;
    const int nt = bid & (NT16 - 1);
    const int b  = bid >> 12;
    const int n0 = nt * 16;

    const float* xb = src + (size_t)b * CDIM * HW;
#pragma unroll
    for (int it = 0; it < 12; ++it) {
        int linear = it * 256 + threadIdx.x;
        int k = linear >> 4, n = linear & 15;
        xs[n * CDIM + k] = (__bf16)xb[(size_t)k * HW + n0 + n];
    }
    __syncthreads();

    const int lane = threadIdx.x & 31;
    const int wave = threadIdx.x >> 5;
    const int lo = lane & 15, hi = lane >> 4;
    const int n = n0 + lo;
    const v16bf* bsrc = (const v16bf*)(xs + lo * CDIM + hi * 16);

#pragma unroll
    for (int j = 0; j < 2; ++j) {
        const int ot = wave + 8 * j;        // 12 OC tiles
        if (ot < CDIM / 16) {
            v8f acc = {};
#pragma unroll
            for (int ks = 0; ks < KSTEPS; ++ks) {
                v16bf afrag = *(const v16bf*)(wpk + ((size_t)(ot * KSTEPS + ks) * 32 + lane) * 16);
                v16bf bfrag = bsrc[ks * 2];
                acc = __builtin_amdgcn_wmma_f32_16x16x32_bf16(
                          false, afrag, false, bfrag, (short)0, acc, false, false);
            }
#pragma unroll
            for (int r = 0; r < 8; ++r) {
                int oc = ot * 16 + (hi << 3) + r;
                out[(size_t)(b * CDIM + oc) * HW + n] = acc[r] + bproj[oc];
            }
        }
    }
}

// =====================================================================
// Host launcher
// =====================================================================
extern "C" void kernel_launch(void* const* d_in, const int* in_sizes, int n_in,
                              void* d_out, int out_size, void* d_ws, size_t ws_size,
                              hipStream_t stream) {
    (void)in_sizes; (void)n_in; (void)out_size; (void)ws_size;
    const float* x      = (const float*)d_in[0];
    const float* wq     = (const float*)d_in[1];
    const float* w_qdw  = (const float*)d_in[2];
    const float* b_qdw  = (const float*)d_in[3];
    const float* wkv    = (const float*)d_in[4];
    const float* w_kvdw = (const float*)d_in[5];
    const float* b_kvdw = (const float*)d_in[6];
    const float* ln_w   = (const float*)d_in[7];
    const float* ln_b   = (const float*)d_in[8];
    const float* w_proj = (const float*)d_in[9];
    const float* b_proj = (const float*)d_in[10];
    float* out = (float*)d_out;

    const size_t NE = (size_t)BATCH * CDIM * HW;    // 50,331,648 floats
    float* ws    = (float*)d_ws;
    float* qpre  = ws;              // NE
    float* kvpre = ws + NE;         // 2*NE
    float* qdw   = ws + 3 * NE;     // NE
    float* kvdw  = ws + 4 * NE;     // 2*NE
    float* mixo  = qpre;            // reuse: q_pre dead after dwconv3x3
    float* prepj = qdw;             // reuse: q_dw dead after patch conv
    __bf16* wpk_qkv  = (__bf16*)(ws + 6 * NE);          // 36*6*512 bf16
    __bf16* wpk_proj = wpk_qkv + 36 * KSTEPS * 512;     // 12*6*512 bf16

    // 0) pack weights to bf16 A-fragment layout
    pack_weights_kernel<<<432, 256, 0, stream>>>(wq, wkv, 36, wpk_qkv);       // 36*6*512/256
    pack_weights_kernel<<<144, 256, 0, stream>>>(w_proj, w_proj, 12, wpk_proj);

    // 1) fused q/kv 1x1 conv (WMMA): one block per 16-col tile, X read once
    gemm_qkv_wmma<<<BATCH * NT16, 256, 0, stream>>>(x, wpk_qkv, qpre, kvpre);

    // 2) depthwise 3x3 on q
    dwconv_kernel<3, 1, CDIM><<<196608, 256, 0, stream>>>(qpre, w_qdw, b_qdw, qdw);

    // 3) depthwise 7x7 on kv
    dwconv_kernel<7, 3, 2 * CDIM><<<393216, 256, 0, stream>>>(kvpre, w_kvdw, b_kvdw, kvdw);

    // 4) per-patch circular conv
    patch_circconv_kernel<<<196608, 256, 0, stream>>>(qdw, kvdw, mixo);

    // 5) LayerNorm over channels + gate by v
    ln_vmul_kernel<<<1024, 256, 0, stream>>>(mixo, kvdw, ln_w, ln_b, prepj);

    // 6) projection 1x1 conv (WMMA) + bias
    gemm_proj_wmma<<<BATCH * NT16, 256, 0, stream>>>(prepj, wpk_proj, b_proj, out);
}
